// ChannelProcessing_54391465837197
// MI455X (gfx1250) — matrix operand order repaired and checked by reading, not compiled, not run on hardware
//
#include <hip/hip_runtime.h>
#include <hip/hip_bf16.h>
#include <math.h>

// ---------------- problem constants (from reference) ----------------
#define BATCH 8
#define NSEQ  3136         // 56*56 tokens
#define TOK   (BATCH*NSEQ) // 25088
#define HH    56
#define WW    56
#define CDIM  768
#define NH    8
#define CHD   96           // CDIM / NH
#define HID   3072
#define LN_EPS 1e-5f

typedef __attribute__((ext_vector_type(16))) __bf16 v16bf;
typedef __attribute__((ext_vector_type(8)))  float  v8f;

union FragBF { v16bf v; uint4 q[2]; };

// -------- CDNA5 async global->LDS copy (no VGPR round trip, ASYNCcnt) --------
#if __has_builtin(__builtin_amdgcn_global_load_async_to_lds_b128)
#define USE_ASYNC_LDS 1
#else
#define USE_ASYNC_LDS 0
#endif

#if USE_ASYNC_LDS
typedef int v4i __attribute__((ext_vector_type(4)));
typedef __attribute__((address_space(1))) v4i gv4i;   // global int4
typedef __attribute__((address_space(3))) v4i lv4i;   // LDS int4
__device__ __forceinline__ void async_cp16(void* lds, const void* g) {
  __builtin_amdgcn_global_load_async_to_lds_b128((gv4i*)g, (lv4i*)lds, 0, 0);
}
#endif

template <int N>
__device__ __forceinline__ void wait_asynccnt() {
#if __has_builtin(__builtin_amdgcn_s_wait_asynccnt)
  __builtin_amdgcn_s_wait_asynccnt(N);
#else
  asm volatile("s_wait_asynccnt %0" :: "i"(N) : "memory");
#endif
}

// ---------------- f32 -> bf16 conversion ----------------
__global__ void cvt_f32_bf16(const float* __restrict__ src, __bf16* __restrict__ dst, int n) {
  int i = blockIdx.x * blockDim.x + threadIdx.x;
  if (i < n) dst[i] = (__bf16)src[i];
}

// ---------------- BF16 WMMA GEMM:  C[M,N] = A[M,K] * W[N,K]^T (+bias) ----------------
// block tile 128x128, 256 threads = 8 waves (2 M x 4 N), wave tile 64x32.
// K-chunk 64, double-buffered LDS, async global->LDS streaming.
// LDS rows padded to 72 bf16 (144B, 16B-aligned) to spread banks.
#define LDK   64
#define LDP   72
__global__ __launch_bounds__(256) void gemm_bf16_nt(
    const __bf16* __restrict__ A,    // [M,K] row-major
    const __bf16* __restrict__ W,    // [N,K] row-major (i.e. B^T with K contiguous)
    const float*  __restrict__ bias, // [N] or nullptr
    float* __restrict__ Cout,        // [M,N] row-major, f32
    int M, int Nn, int K)
{
  __shared__ __bf16 As[2][128 * LDP];
  __shared__ __bf16 Bs[2][128 * LDP];

  const int nblk = Nn >> 7;
  const int m0 = (blockIdx.x / nblk) << 7;
  const int n0 = (blockIdx.x % nblk) << 7;
  const int tid  = threadIdx.x;
  const int lane = tid & 31;
  const int wm = (tid >> 5) >> 2;   // 0..1  (wave M)
  const int wn = (tid >> 5) & 3;    // 0..3  (wave N)
  const int lr = lane & 15;
  const int lh = lane >> 4;

  v8f acc[4][2] = {};

  // cooperative copy mapping: 8 threads/row (8-elt segs), 32 rows/pass, 4 passes
  const int crow = tid >> 3;          // 0..31
  const int cseg = (tid & 7) << 3;    // 0,8,...,56 (bf16 elements)
  const __bf16* Ab = A + (size_t)m0 * K;
  const __bf16* Wb = W + (size_t)n0 * K;

  const int nIter = K >> 6;           // K is a multiple of 64 here (768 / 3072)

#if USE_ASYNC_LDS
  // issue one 128x64 A-tile + B-tile: 8 async b128 ops per thread
  #define COPY_TILES(buf, kof)                                                      \
    {                                                                               \
      _Pragma("unroll")                                                             \
      for (int rr = 0; rr < 4; ++rr) {                                              \
        const int row = crow + rr * 32;                                             \
        async_cp16(&As[buf][row * LDP + cseg], Ab + (size_t)row * K + (kof) + cseg);\
        async_cp16(&Bs[buf][row * LDP + cseg], Wb + (size_t)row * K + (kof) + cseg);\
      }                                                                             \
    }
  COPY_TILES(0, 0)
  COPY_TILES(1, LDK)
#else
  #define COPY_TILES(buf, kof)                                                      \
    {                                                                               \
      uint4 ta[4], tb[4];                                                           \
      _Pragma("unroll")                                                             \
      for (int rr = 0; rr < 4; ++rr) {                                              \
        const int row = crow + rr * 32;                                             \
        ta[rr] = *(const uint4*)(Ab + (size_t)row * K + (kof) + cseg);              \
        tb[rr] = *(const uint4*)(Wb + (size_t)row * K + (kof) + cseg);              \
      }                                                                             \
      _Pragma("unroll")                                                             \
      for (int rr = 0; rr < 4; ++rr) {                                              \
        const int row = crow + rr * 32;                                             \
        *(uint4*)&As[buf][row * LDP + cseg] = ta[rr];                               \
        *(uint4*)&Bs[buf][row * LDP + cseg] = tb[rr];                               \
      }                                                                             \
    }
  COPY_TILES(0, 0)
  __syncthreads();
#endif

  for (int i = 0, k0 = 0; i < nIter; ++i, k0 += LDK) {
    const int cur = i & 1;
#if USE_ASYNC_LDS
    if (i + 1 < nIter) wait_asynccnt<8>();  // current chunk landed; next still in flight
    else               wait_asynccnt<0>();
    __syncthreads();
#else
    if (i + 1 < nIter) COPY_TILES(1 - cur, k0 + LDK)  // overlap next-tile fetch w/ math
#endif

    const __bf16* as = As[cur];
    const __bf16* bs = Bs[cur];
#pragma unroll
    for (int kk = 0; kk < 2; ++kk) {
      const int ko = kk * 32;
      // B frags: lane<16 -> col=lr, K ko+0..15 ; lane>=16 -> K ko+16..31
      FragBF bfrag[2];
#pragma unroll
      for (int ni = 0; ni < 2; ++ni) {
        const __bf16* p = &bs[(wn * 32 + ni * 16 + lr) * LDP + ko + lh * 16];
        bfrag[ni].q[0] = *(const uint4*)p;
        bfrag[ni].q[1] = *(const uint4*)(p + 8);
      }
      // A frags: lane<16 -> row=lr, K ko+0..15 ; lane>=16 -> K ko+16..31
#pragma unroll
      for (int mi = 0; mi < 4; ++mi) {
        FragBF afrag;
        const __bf16* p = &as[(wm * 64 + mi * 16 + lr) * LDP + ko + lh * 16];
        afrag.q[0] = *(const uint4*)p;
        afrag.q[1] = *(const uint4*)(p + 8);
#pragma unroll
        for (int ni = 0; ni < 2; ++ni) {
          acc[mi][ni] = __builtin_amdgcn_wmma_f32_16x16x32_bf16(
              false, afrag.v, false, bfrag[ni].v, (short)0, acc[mi][ni], false, false);
        }
      }
    }
    __syncthreads();
#if USE_ASYNC_LDS
    if (i + 2 < nIter) COPY_TILES(cur, k0 + 2 * LDK)  // refill freed buffer
#endif
  }
#undef COPY_TILES

  // epilogue: C/D layout -> lane<16: rows +0..7 ; lane>=16: rows +8..15 ; col = lr
#pragma unroll
  for (int ni = 0; ni < 2; ++ni) {
    const int col = n0 + wn * 32 + ni * 16 + lr;
    const float bv = bias ? bias[col] : 0.0f;
#pragma unroll
    for (int mi = 0; mi < 4; ++mi) {
      const int rowb = m0 + wm * 64 + mi * 16 + lh * 8;
#pragma unroll
      for (int r = 0; r < 8; ++r) {
        Cout[(size_t)(rowb + r) * Nn + col] = acc[mi][ni][r] + bv;
      }
    }
  }
}

// ---------------- column softmax stats over the token axis (online max/sum) ----------
__global__ __launch_bounds__(256) void colstats_kernel(
    const float* __restrict__ P, float* __restrict__ mOut, float* __restrict__ sOut)
{
  __shared__ float sm[8][32], ss[8][32];
  const int tx = threadIdx.x, ty = threadIdx.y;
  const int b = blockIdx.x / (CDIM / 32);
  const int c = (blockIdx.x % (CDIM / 32)) * 32 + tx;
  float m = -INFINITY, s = 0.0f;
  for (int n = ty; n < NSEQ; n += 8) {
    float v = P[((size_t)b * NSEQ + n) * CDIM + c];
    if (v > m) { s = s * expf(m - v) + 1.0f; m = v; }
    else       { s += expf(v - m); }
  }
  sm[ty][tx] = m; ss[ty][tx] = s;
  __syncthreads();
  if (ty == 0) {
    for (int j = 1; j < 8; ++j) {
      float m2 = sm[j][tx], s2 = ss[j][tx];
      float Mx = fmaxf(m, m2);
      s = s * expf(m - Mx) + s2 * expf(m2 - Mx);
      m = Mx;
    }
    mOut[b * CDIM + c] = m;
    sOut[b * CDIM + c] = s;
  }
}

// ---------------- kpool[b,h,n] = mean_ch softmax_tokens(x)[b,n,h,:] ----------------
__global__ void kpool_kernel(const float* __restrict__ x,
                             const float* __restrict__ kmax,
                             const float* __restrict__ ksum,
                             float* __restrict__ kpool)
{
  int g = blockIdx.x * blockDim.x + threadIdx.x;
  if (g >= BATCH * NSEQ * NH) return;
  int h = g % NH;
  int n = (g / NH) % NSEQ;
  int b = g / (NH * NSEQ);
  const float* xr = x + ((size_t)b * NSEQ + n) * CDIM + h * CHD;
  const float* mr = kmax + b * CDIM + h * CHD;
  const float* sr = ksum + b * CDIM + h * CHD;
  float acc = 0.0f;
  for (int ci = 0; ci < CHD; ++ci)
    acc += expf(xr[ci] - mr[ci]) / sr[ci];
  kpool[((size_t)b * NH + h) * NSEQ + n] = acc * (1.0f / CHD);
}

// ---------------- attn[b,c] = sigmoid( sum_n softmax(Q)[b,n,c] * kpool[b,h,n] ) * T[h]
__global__ __launch_bounds__(256) void attn_kernel(
    const float* __restrict__ Q, const float* __restrict__ qmax,
    const float* __restrict__ qsum, const float* __restrict__ kpool,
    const float* __restrict__ temp, float* __restrict__ attn)
{
  __shared__ float red[8][32];
  const int tx = threadIdx.x, ty = threadIdx.y;
  const int b = blockIdx.x / (CDIM / 32);
  const int c = (blockIdx.x % (CDIM / 32)) * 32 + tx;
  const int h = c / CHD;   // 96 = 3*32 -> whole 32-tile in one head
  const float m   = qmax[b * CDIM + c];
  const float inv = 1.0f / qsum[b * CDIM + c];
  const float* kp = kpool + ((size_t)b * NH + h) * NSEQ;
  float acc = 0.0f;
  for (int n = ty; n < NSEQ; n += 8)
    acc += expf(Q[((size_t)b * NSEQ + n) * CDIM + c] - m) * inv * kp[n];
  red[ty][tx] = acc;
  __syncthreads();
  if (ty == 0) {
    for (int j = 1; j < 8; ++j) acc += red[j][tx];
    attn[b * CDIM + c] = temp[h] / (1.0f + expf(-acc));
  }
}

// ---------------- depthwise 3x3 conv (SAME) + bias + exact GELU, f32 -> bf16 --------
__global__ __launch_bounds__(256) void dwconv_gelu_kernel(
    const float* __restrict__ Hin, const float* __restrict__ w,
    const float* __restrict__ bias, __bf16* __restrict__ G)
{
  const int cchunk = blockIdx.x % (HID / 256);
  const int sp = blockIdx.x / (HID / 256);
  const int x0 = sp % WW;
  const int y0 = (sp / WW) % HH;
  const int b  = sp / (WW * HH);
  const int c  = cchunk * 256 + threadIdx.x;
  float acc = bias[c];
  const float* wc = w + c * 9;
#pragma unroll
  for (int ky = 0; ky < 3; ++ky) {
    int y = y0 + ky - 1;
    if (y < 0 || y >= HH) continue;
#pragma unroll
    for (int kx = 0; kx < 3; ++kx) {
      int x = x0 + kx - 1;
      if (x < 0 || x >= WW) continue;
      acc += Hin[((size_t)b * NSEQ + y * WW + x) * HID + c] * wc[ky * 3 + kx];
    }
  }
  float g = 0.5f * acc * (1.0f + erff(acc * 0.70710678118654752f));
  G[((size_t)b * NSEQ + y0 * WW + x0) * HID + c] = (__bf16)g;
}

// ---------------- LayerNorm over C + attn scaling; writes out and av layouts -------
__global__ __launch_bounds__(256) void ln_out_kernel(
    const float* __restrict__ H2, const float* __restrict__ attn,
    const float* __restrict__ lnw, const float* __restrict__ lnb,
    float* __restrict__ out, float* __restrict__ av)
{
  __shared__ float rs[256], rq[256];
  const int t = blockIdx.x;              // token index
  const int b = t / NSEQ, n = t % NSEQ;
  const int tid = threadIdx.x;
  float v0 = H2[(size_t)t * CDIM + tid];
  float v1 = H2[(size_t)t * CDIM + tid + 256];
  float v2 = H2[(size_t)t * CDIM + tid + 512];
  rs[tid] = v0 + v1 + v2;
  rq[tid] = v0 * v0 + v1 * v1 + v2 * v2;
  __syncthreads();
  for (int off = 128; off > 0; off >>= 1) {
    if (tid < off) { rs[tid] += rs[tid + off]; rq[tid] += rq[tid + off]; }
    __syncthreads();
  }
  const float mean = rs[0] * (1.0f / CDIM);
  const float var  = rq[0] * (1.0f / CDIM) - mean * mean;
  const float rstd = rsqrtf(var + LN_EPS);
  const float* at = attn + b * CDIM;
  float vv[3] = { v0, v1, v2 };
#pragma unroll
  for (int kk = 0; kk < 3; ++kk) {
    int c = tid + kk * 256;
    float y = (vv[kk] - mean) * rstd * lnw[c] + lnb[c];
    float o = at[c] * y;
    out[(size_t)t * CDIM + c] = o;                 // out[b,n,c]
    av[((size_t)b * CDIM + c) * NSEQ + n] = o;     // av[b,h,ci,n] == [(b*C+c)*N+n]
  }
}

// =====================================================================
extern "C" void kernel_launch(void* const* d_in, const int* in_sizes, int n_in,
                              void* d_out, int out_size, void* d_ws, size_t ws_size,
                              hipStream_t stream) {
  (void)in_sizes; (void)n_in; (void)out_size; (void)ws_size;
  const float* x     = (const float*)d_in[0];
  // d_in[1]=H, d_in[2]=W (ints, fixed 56x56)
  const float* q_w   = (const float*)d_in[3];
  const float* temp  = (const float*)d_in[4];
  const float* fc1_w = (const float*)d_in[5];
  const float* fc1_b = (const float*)d_in[6];
  const float* dw_w  = (const float*)d_in[7];
  const float* dw_b  = (const float*)d_in[8];
  const float* fc2_w = (const float*)d_in[9];
  const float* fc2_b = (const float*)d_in[10];
  const float* ln_w  = (const float*)d_in[11];
  const float* ln_b  = (const float*)d_in[12];

  // ---- workspace carving (256B aligned) ----
  char* ws = (char*)d_ws;
  auto carve = [&](size_t bytes) -> char* {
    char* p = ws; ws += (bytes + 255) & ~(size_t)255; return p;
  };
  __bf16* xb   = (__bf16*)carve((size_t)TOK * CDIM * 2);
  __bf16* qwb  = (__bf16*)carve((size_t)CDIM * CDIM * 2);
  __bf16* w1b  = (__bf16*)carve((size_t)HID * CDIM * 2);
  __bf16* w2b  = (__bf16*)carve((size_t)CDIM * HID * 2);
  float*  Qf   = (float*) carve((size_t)TOK * CDIM * 4);
  float*  H1   = (float*) carve((size_t)TOK * HID * 4);
  __bf16* G    = (__bf16*)carve((size_t)TOK * HID * 2);
  float*  kmax = (float*) carve((size_t)BATCH * CDIM * 4);
  float*  ksum = (float*) carve((size_t)BATCH * CDIM * 4);
  float*  qmax = (float*) carve((size_t)BATCH * CDIM * 4);
  float*  qsum = (float*) carve((size_t)BATCH * CDIM * 4);
  float*  kpool= (float*) carve((size_t)BATCH * NH * NSEQ * 4);
  float*  attn = (float*) carve((size_t)BATCH * CDIM * 4);

  float* outp = (float*)d_out;                        // [B,N,C]
  float* avp  = outp + (size_t)TOK * CDIM;            // [B,nh,ch,N]
  float* H2   = outp;                                 // fc2 output staged in d_out 1st half

  // ---- bf16 conversions ----
  {
    int n1 = TOK * CDIM;
    cvt_f32_bf16<<<(n1 + 255) / 256, 256, 0, stream>>>(x, xb, n1);
    int n2 = CDIM * CDIM;
    cvt_f32_bf16<<<(n2 + 255) / 256, 256, 0, stream>>>(q_w, qwb, n2);
    int n3 = HID * CDIM;
    cvt_f32_bf16<<<(n3 + 255) / 256, 256, 0, stream>>>(fc1_w, w1b, n3);
    cvt_f32_bf16<<<(n3 + 255) / 256, 256, 0, stream>>>(fc2_w, w2b, n3);
  }

  // ---- WMMA GEMMs ----
  gemm_bf16_nt<<<(TOK / 128) * (CDIM / 128), 256, 0, stream>>>(
      xb, qwb, nullptr, Qf, TOK, CDIM, CDIM);          // Q = x @ q_w^T
  gemm_bf16_nt<<<(TOK / 128) * (HID / 128), 256, 0, stream>>>(
      xb, w1b, fc1_b, H1, TOK, HID, CDIM);             // H1 = x @ fc1_w^T + b

  // ---- channel-attention branch ----
  colstats_kernel<<<BATCH * (CDIM / 32), dim3(32, 8), 0, stream>>>(x, kmax, ksum);
  colstats_kernel<<<BATCH * (CDIM / 32), dim3(32, 8), 0, stream>>>(Qf, qmax, qsum);
  {
    int n = BATCH * NSEQ * NH;
    kpool_kernel<<<(n + 255) / 256, 256, 0, stream>>>(x, kmax, ksum, kpool);
  }
  attn_kernel<<<BATCH * (CDIM / 32), dim3(32, 8), 0, stream>>>(
      Qf, qmax, qsum, kpool, temp, attn);

  // ---- value branch: dwconv + GELU, fc2, LN + scale ----
  dwconv_gelu_kernel<<<BATCH * HH * WW * (HID / 256), 256, 0, stream>>>(
      H1, dw_w, dw_b, G);
  gemm_bf16_nt<<<(TOK / 128) * (CDIM / 128), 256, 0, stream>>>(
      G, w2b, fc2_b, H2, TOK, CDIM, HID);              // H2 = G @ fc2_w^T + b
  ln_out_kernel<<<TOK, 256, 0, stream>>>(H2, attn, ln_w, ln_b, outp, avp);
}
